// EncoderLayer_41850161332866
// MI455X (gfx1250) — matrix operand order repaired and checked by reading, not compiled, not run on hardware
//
#include <hip/hip_runtime.h>

// ---------------------------------------------------------------------------
// MI455X (gfx1250) encoder layer.
// Roofline: attn output write = 537 MB -> ~23us floor @ 23.3 TB/s.
// Compute ~40 GFLOP via v_wmma_f32_16x16x32_f16 (f16 in, f32 accum) -> tiny.
// 2 v_exp per attn element via 3-pass (max / sumexp / write) softmax.
// Mask pre-packed to a 4 MB bitmask (L2-resident): no barriers / no LDS mask
// traffic in the attention passes. K/V (~25 MB) are L2-resident so the QK^T
// recompute across passes costs no HBM traffic.
// uQ.uK^T folded into the head GEMM by augmenting head dim to 128.
// ---------------------------------------------------------------------------

typedef __attribute__((ext_vector_type(16))) _Float16 v16h;
typedef __attribute__((ext_vector_type(8)))  _Float16 h8;
typedef __attribute__((ext_vector_type(8)))  float    v8f;

constexpr int B = 8, L = 2048, D = 64, H = 4;

#define LOG2E 1.4426950408889634f
constexpr float SCALE2 = 0.125f * LOG2E;     // (1/sqrt(64)) * log2(e)
constexpr float MASKV2 = -1e9f * LOG2E;      // masked score, base-2 domain

__device__ __forceinline__ v8f wmma16(v16h a, v16h b, v8f c) {
  // D = A(16x32 f16) * B(32x16 f16) + C(16x16 f32)
  return __builtin_amdgcn_wmma_f32_16x16x32_f16(false, a, false, b,
                                                (short)0, c, false, false);
}

// Build a 16-half fragment from two contiguous 8-half (16B) global loads.
__device__ __forceinline__ v16h ld_frag(const _Float16* p0, const _Float16* p1) {
  h8 lo = *(const h8*)p0;
  h8 hi = *(const h8*)p1;
  v16h r;
#pragma unroll
  for (int i = 0; i < 8; ++i) { r[i] = lo[i]; r[i + 8] = hi[i]; }
  return r;
}

// ---------------------------------------------------------------------------
// Kernel 1: per-batch scalar gates S_q, S_k, S_v (tiny).
// ---------------------------------------------------------------------------
__global__ void gates_kernel(const float* __restrict__ u,
                             const float* __restrict__ sq1, const float* __restrict__ sq2,
                             const float* __restrict__ sk1, const float* __restrict__ sk2,
                             const float* __restrict__ sv1, const float* __restrict__ sv2,
                             float* __restrict__ S) {
  int t = threadIdx.x;
  if (t >= B * 3) return;
  int b = t / 3, g = t % 3;
  const float* s1 = (g == 0) ? sq1 : (g == 1) ? sk1 : sv1;
  const float* s2 = (g == 0) ? sq2 : (g == 1) ? sk2 : sv2;
  float acc = 0.f;
  for (int j = 0; j < 32; ++j) {
    float hsum = 0.f;
    for (int c = 0; c < 64; ++c) hsum += u[b * 64 + c] * s1[c * 32 + j];
    acc += fmaxf(hsum, 0.f) * s2[j];
  }
  S[t] = acc;  // S[b*3 + g]
}

// ---------------------------------------------------------------------------
// Kernel 2: pack bool mask [B,L,L] (1 byte/elem) -> bitmask (1 bit/elem, 4 MB).
// Bit i of word w == mask byte (w*32 + i). Rows are L=2048 bits, so a 16x16
// score tile's mask row is a single aligned u16.
// ---------------------------------------------------------------------------
__global__ void __launch_bounds__(256)
pack_mask_kernel(const unsigned char* __restrict__ mask,
                 unsigned int* __restrict__ pk) {
  size_t w = (size_t)blockIdx.x * 256 + threadIdx.x;   // word index
  const uint4* p4 = (const uint4*)(mask + w * 32);
  uint4 a = p4[0], bq = p4[1];
  unsigned int src[8] = {a.x, a.y, a.z, a.w, bq.x, bq.y, bq.z, bq.w};
  unsigned int bits = 0;
#pragma unroll
  for (int j = 0; j < 8; ++j) {
    unsigned int v = src[j] & 0x01010101u;            // bool bytes -> bit0 of each
    bits |= ((v | (v >> 7) | (v >> 14) | (v >> 21)) & 0xFu) << (4 * j);
  }
  pk[w] = bits;
}

// ---------------------------------------------------------------------------
// Kernel 3: projections -> f16 staged tensors for WMMA attention.
//   Qa [B,H,L,128] row-major : [Q_h | S_q*uQ]
//   Kt [B,H,128,L] feat-major: [K_h | S_k*uK]   (B-frag rows contiguous in L)
//   Va [B,H,L,64]  row-major : V_h + S_v*uV
// ---------------------------------------------------------------------------
__global__ void __launch_bounds__(256)
proj_kernel(const float* __restrict__ x,
            const float* __restrict__ wq, const float* __restrict__ wk,
            const float* __restrict__ wv,
            const float* __restrict__ uqw, const float* __restrict__ ukw,
            const float* __restrict__ uvw,
            const float* __restrict__ S,
            _Float16* __restrict__ Qa, _Float16* __restrict__ Kt,
            _Float16* __restrict__ Va) {
  __shared__ float sx[64][65];  // padded: bank-conflict free
  const int tid = threadIdx.x;
  const int b   = blockIdx.x >> 5;           // 32 tiles of 64 rows per batch
  const int l0  = (blockIdx.x & 31) * 64;
#pragma unroll
  for (int i = 0; i < 16; ++i) {
    int idx = tid + i * 256;
    sx[idx >> 6][idx & 63] = x[((size_t)b * L + l0 + (idx >> 6)) * 64 + (idx & 63)];
  }
  __syncthreads();
  const int r = tid & 63, h = tid >> 6;      // thread = (row, head)
  const size_t l  = (size_t)l0 + r;
  const size_t bh = (size_t)b * H + h;
  const float Sq = S[b * 3 + 0], Sk = S[b * 3 + 1], Sv = S[b * 3 + 2];
  float xr[64];
#pragma unroll
  for (int c = 0; c < 64; ++c) xr[c] = sx[r][c];
  for (int d = 0; d < 64; ++d) {
    float q = 0, k = 0, v = 0, uq = 0, uk = 0, uv = 0;
#pragma unroll
    for (int c = 0; c < 64; ++c) {
      float xv = xr[c];
      q  += xv * wq[c * 256 + h * 64 + d];
      k  += xv * wk[c * 256 + h * 64 + d];
      v  += xv * wv[c * 256 + h * 64 + d];
      uq += xv * uqw[c * 64 + d];
      uk += xv * ukw[c * 64 + d];
      uv += xv * uvw[c * 64 + d];
    }
    Qa[(bh * L + l) * 128 + d]       = (_Float16)q;
    Qa[(bh * L + l) * 128 + 64 + d]  = (_Float16)(uq * Sq);
    Kt[(bh * 128 + d) * L + l]       = (_Float16)k;
    Kt[(bh * 128 + 64 + d) * L + l]  = (_Float16)(uk * Sk);
    Va[(bh * L + l) * 64 + d]        = (_Float16)(v + uv * Sv);
  }
}

// ---------------------------------------------------------------------------
// Kernel 4: fused attention. Block = 8 independent waves; wave -> (head = w&3,
// qhalf = w>>2), 16 queries/wave. 3 passes over K (max / sumexp / write+PV),
// all WMMA. No barriers: mask comes from the packed bitmask via one broadcast
// u16 load per (row, 16-key tile).
// ---------------------------------------------------------------------------
__global__ void __launch_bounds__(256)
attn_kernel(const _Float16* __restrict__ Qa, const _Float16* __restrict__ Kt,
            const _Float16* __restrict__ Va, const unsigned short* __restrict__ mpk,
            float* __restrict__ attn, float* __restrict__ Ctx) {
  __shared__ _Float16 sP[8][16][48];          // per-wave prob tile, 16B-aligned rows

  const int tid  = threadIdx.x;
  const int wave = tid >> 5, lane = tid & 31;
  const int ln16 = lane & 15, hl = lane >> 4; // C-layout: row = j + 8*hl, col = ln16
  const int h = wave & 3, qh = wave >> 2;
  const int b  = blockIdx.x >> 6;             // 64 q-tiles of 32 per batch
  const int qt = blockIdx.x & 63;
  const int q0 = qt * 32 + qh * 16;
  const size_t bh = (size_t)b * H + h;
  const _Float16* QaBH = Qa + bh * (size_t)L * 128;
  const _Float16* KtBH = Kt + bh * (size_t)128 * L;
  const _Float16* VaBH = Va + bh * (size_t)L * 64;

  // Per-j mask row pointers (u16 per 16-key tile; row stride = L/16 = 128 u16).
  const unsigned short* mrow[8];
#pragma unroll
  for (int j = 0; j < 8; ++j)
    mrow[j] = mpk + ((size_t)b * L + q0 + j + 8 * hl) * 128;

  // Resident A-fragments of augmented Q (16 x 128 = 4 chunks of 16x32).
  // A-layout (16-bit): lanes<16 hold K {0..7,16..23}, lanes>=16 hold {8..15,24..31}.
  v16h aq[4];
#pragma unroll
  for (int f = 0; f < 4; ++f) {
    const _Float16* p = QaBH + (size_t)(q0 + ln16) * 128 + f * 32 + hl * 8;
    aq[f] = ld_frag(p, p + 16);
  }

  auto score_tile = [&](int k0) -> v8f {
    v8f c = {};
#pragma unroll
    for (int f = 0; f < 4; ++f) {
      // B-layout (16-bit): lane%16 = N column, K rows {8hl..8hl+7, 16+8hl..16+8hl+7}.
      const _Float16* p = KtBH + (size_t)(f * 32 + hl * 8) * L + k0 + ln16;
      v16h kb;
#pragma unroll
      for (int i = 0; i < 8; ++i) { kb[i] = p[(size_t)i * L]; kb[i + 8] = p[(size_t)(i + 16) * L]; }
      c = wmma16(aq[f], kb, c);
    }
    return c;
  };

  // ---- pass A: row max (no transcendentals) ----
  float m[8];
#pragma unroll
  for (int j = 0; j < 8; ++j) m[j] = -3.0e38f;
  for (int kt = 0; kt < 128; ++kt) {
    int k0 = kt * 16;
    v8f c = score_tile(k0);
#pragma unroll
    for (int j = 0; j < 8; ++j) {
      float s = ((mrow[j][kt] >> ln16) & 1) ? MASKV2 : c[j] * SCALE2;
      m[j] = fmaxf(m[j], s);
    }
  }
#pragma unroll
  for (int j = 0; j < 8; ++j) {
    m[j] = fmaxf(m[j], __shfl_xor(m[j], 8, 32));
    m[j] = fmaxf(m[j], __shfl_xor(m[j], 4, 32));
    m[j] = fmaxf(m[j], __shfl_xor(m[j], 2, 32));
    m[j] = fmaxf(m[j], __shfl_xor(m[j], 1, 32));
  }

  // ---- pass B: row sum of exp2 (1 v_exp per element) ----
  float lsum[8];
#pragma unroll
  for (int j = 0; j < 8; ++j) lsum[j] = 0.f;
  for (int kt = 0; kt < 128; ++kt) {
    int k0 = kt * 16;
    v8f c = score_tile(k0);
#pragma unroll
    for (int j = 0; j < 8; ++j) {
      float s = ((mrow[j][kt] >> ln16) & 1) ? MASKV2 : c[j] * SCALE2;
      lsum[j] += exp2f(s - m[j]);
    }
  }
  float il[8];
#pragma unroll
  for (int j = 0; j < 8; ++j) {
    lsum[j] += __shfl_xor(lsum[j], 8, 32);
    lsum[j] += __shfl_xor(lsum[j], 4, 32);
    lsum[j] += __shfl_xor(lsum[j], 2, 32);
    lsum[j] += __shfl_xor(lsum[j], 1, 32);
    il[j] = 1.0f / lsum[j];
  }

  // ---- pass C: write normalized attn (NT stores) + P@V via WMMA ----
  v8f cacc[4];
#pragma unroll
  for (int t = 0; t < 4; ++t) cacc[t] = (v8f){};
  for (int kt2 = 0; kt2 < 64; ++kt2) {
    // Hoist V B-fragments: their loads overlap the two score sub-tiles below.
    v16h vb[4];
#pragma unroll
    for (int t = 0; t < 4; ++t) {
      const _Float16* vp = VaBH + (size_t)(kt2 * 32 + hl * 8) * 64 + t * 16 + ln16;
#pragma unroll
      for (int i = 0; i < 8; ++i) { vb[t][i] = vp[i * 64]; vb[t][i + 8] = vp[(i + 16) * 64]; }
    }
#pragma unroll
    for (int sub = 0; sub < 2; ++sub) {
      int kt = kt2 * 2 + sub, k0 = kt * 16;
      v8f c = score_tile(k0);
#pragma unroll
      for (int j = 0; j < 8; ++j) {
        float s = ((mrow[j][kt] >> ln16) & 1) ? MASKV2 : c[j] * SCALE2;
        float p = exp2f(s - m[j]) * il[j];
        __builtin_nontemporal_store(
            p, &attn[(bh * L + q0 + j + 8 * hl) * (size_t)L + k0 + ln16]);
        sP[wave][j + 8 * hl][sub * 16 + ln16] = (_Float16)p;   // per-wave tile
      }
    }
    // Re-shape probs (C-layout -> A-layout) through LDS; per-wave DS ops are
    // in-order so no barrier is required for this private tile.
    v16h pa;
#pragma unroll
    for (int i = 0; i < 8; ++i) {
      pa[i]     = sP[wave][ln16][hl * 8 + i];
      pa[i + 8] = sP[wave][ln16][hl * 8 + 16 + i];
    }
#pragma unroll
    for (int t = 0; t < 4; ++t) cacc[t] = wmma16(pa, vb[t], cacc[t]);
  }
  // ctx [B, L, H*64]
#pragma unroll
  for (int t = 0; t < 4; ++t)
#pragma unroll
    for (int j = 0; j < 8; ++j)
      Ctx[((size_t)b * L + q0 + j + 8 * hl) * 256 + h * 64 + t * 16 + ln16] =
          cacc[t][j];
}

// ---------------------------------------------------------------------------
// Kernel 5: ctx@fcw + residual -> LN1 -> FFN -> residual -> LN2 -> res.
// 128 threads / 32 rows; thread = (row, 16-col group). LDS < 64KB.
// ---------------------------------------------------------------------------
__global__ void __launch_bounds__(128)
epilogue_kernel(const float* __restrict__ Ctx, const float* __restrict__ x,
                const float* __restrict__ fcw,
                const float* __restrict__ g1, const float* __restrict__ b1,
                const float* __restrict__ f1, const float* __restrict__ f2,
                const float* __restrict__ g2, const float* __restrict__ b2,
                float* __restrict__ res) {
  __shared__ float sc[32][257];
  __shared__ float sa[32][65];
  __shared__ float sb[32][65];
  __shared__ float mu[32], rs[32];
  const int tid = threadIdx.x;
  const int b   = blockIdx.x >> 6;          // 64 tiles of 32 rows per batch
  const int l0  = (blockIdx.x & 63) * 32;
#pragma unroll
  for (int i = 0; i < 64; ++i) {
    int idx = tid + i * 128;
    sc[idx >> 8][idx & 255] = Ctx[((size_t)b * L + l0 + (idx >> 8)) * 256 + (idx & 255)];
  }
#pragma unroll
  for (int i = 0; i < 16; ++i) {
    int idx = tid + i * 128;
    sa[idx >> 6][idx & 63] = x[((size_t)b * L + l0 + (idx >> 6)) * 64 + (idx & 63)];
  }
  __syncthreads();
  const int r = tid & 31, g = tid >> 5;
  // FC + residual
  for (int jj = 0; jj < 16; ++jj) {
    int col = g * 16 + jj;
    float o = 0.f;
    for (int c = 0; c < 256; ++c) o += sc[r][c] * fcw[c * 64 + col];
    sb[r][col] = o + sa[r][col];
  }
  __syncthreads();
  if (g == 0) {
    float s = 0.f, s2 = 0.f;
    for (int c = 0; c < 64; ++c) { float v = sb[r][c]; s += v; s2 += v * v; }
    float mn = s * (1.f / 64.f);
    mu[r] = mn; rs[r] = rsqrtf(s2 * (1.f / 64.f) - mn * mn + 1e-5f);
  }
  __syncthreads();
  for (int jj = 0; jj < 16; ++jj) {
    int col = g * 16 + jj;
    sa[r][col] = (sb[r][col] - mu[r]) * rs[r] * g1[col] + b1[col];  // enc
  }
  __syncthreads();
  for (int jj = 0; jj < 16; ++jj) {
    int col = g * 16 + jj;
    float hsum = 0.f;
    for (int c = 0; c < 64; ++c) hsum += sa[r][c] * f1[c * 64 + col];
    sb[r][col] = fmaxf(hsum, 0.f);                                   // relu hidden
  }
  __syncthreads();
  for (int jj = 0; jj < 16; ++jj) {
    int col = g * 16 + jj;
    float o2 = 0.f;
    for (int c = 0; c < 64; ++c) o2 += sb[r][c] * f2[c * 64 + col];
    sc[r][col] = o2 + sa[r][col];                                    // + enc
  }
  __syncthreads();
  if (g == 0) {
    float s = 0.f, s2 = 0.f;
    for (int c = 0; c < 64; ++c) { float v = sc[r][c]; s += v; s2 += v * v; }
    float mn = s * (1.f / 64.f);
    mu[r] = mn; rs[r] = rsqrtf(s2 * (1.f / 64.f) - mn * mn + 1e-5f);
  }
  __syncthreads();
  for (int jj = 0; jj < 16; ++jj) {
    int col = g * 16 + jj;
    res[((size_t)b * L + l0 + r) * 64 + col] =
        (sc[r][col] - mu[r]) * rs[r] * g2[col] + b2[col];
  }
}

// ---------------------------------------------------------------------------
extern "C" void kernel_launch(void* const* d_in, const int* in_sizes, int n_in,
                              void* d_out, int out_size, void* d_ws, size_t ws_size,
                              hipStream_t stream) {
  const float* x            = (const float*)d_in[0];
  const unsigned char* mask = (const unsigned char*)d_in[1];  // jnp.bool_ = 1 byte
  const float* u   = (const float*)d_in[2];
  const float* wq  = (const float*)d_in[3];
  const float* wk  = (const float*)d_in[4];
  const float* wv  = (const float*)d_in[5];
  const float* uqw = (const float*)d_in[6];
  const float* ukw = (const float*)d_in[7];
  const float* uvw = (const float*)d_in[8];
  const float* sq1 = (const float*)d_in[9];
  const float* sq2 = (const float*)d_in[10];
  const float* sk1 = (const float*)d_in[11];
  const float* sk2 = (const float*)d_in[12];
  const float* sv1 = (const float*)d_in[13];
  const float* sv2 = (const float*)d_in[14];
  const float* fcw = (const float*)d_in[15];
  const float* ln1g = (const float*)d_in[16];
  const float* ln1b = (const float*)d_in[17];
  const float* ffn1 = (const float*)d_in[18];
  const float* ffn2 = (const float*)d_in[19];
  const float* ln2g = (const float*)d_in[20];
  const float* ln2b = (const float*)d_in[21];

  float* res  = (float*)d_out;                    // [B,L,64]
  float* attn = res + (size_t)B * L * D;          // [B,H,L,L]

  // workspace layout (~62.9 MB)
  char* ws = (char*)d_ws;
  float* S      = (float*)ws;                              // 24 gates
  _Float16* Qa  = (_Float16*)(ws + 256);                   // [B,H,L,128] f16
  _Float16* Kt  = Qa + (size_t)B * H * L * 128;            // [B,H,128,L] f16
  _Float16* Va  = Kt + (size_t)B * H * L * 128;            // [B,H,L,64]  f16
  float*    Ctx = (float*)(Va + (size_t)B * H * L * 64);   // [B,L,256]   f32
  unsigned int* Mpk = (unsigned int*)(Ctx + (size_t)B * L * 256);  // bitmask 4 MB

  gates_kernel<<<1, 32, 0, stream>>>(u, sq1, sq2, sk1, sk2, sv1, sv2, S);
  pack_mask_kernel<<<((size_t)B * L * L / 32) / 256, 256, 0, stream>>>(mask, Mpk);
  proj_kernel<<<B * (L / 64), 256, 0, stream>>>(x, wq, wk, wv, uqw, ukw, uvw,
                                                S, Qa, Kt, Va);
  attn_kernel<<<B * (L / 32), 256, 0, stream>>>(Qa, Kt, Va,
                                                (const unsigned short*)Mpk,
                                                attn, Ctx);
  epilogue_kernel<<<B * (L / 32), 128, 0, stream>>>(Ctx, x, fcw, ln1g, ln1b,
                                                    ffn1, ffn2, ln2g, ln2b, res);
}
